// Attention_10282151707309
// MI455X (gfx1250) — matrix operand order repaired and checked by reading, not compiled, not run on hardware
//
#include <hip/hip_runtime.h>
#include <hip/hip_bf16.h>

typedef _Float16 f16;
typedef __attribute__((ext_vector_type(16))) _Float16 v16h;
typedef __attribute__((ext_vector_type(8)))  _Float16 v8h;
typedef __attribute__((ext_vector_type(8)))  float    v8f;
typedef unsigned int v4u __attribute__((ext_vector_type(4)));
typedef int          v8i __attribute__((ext_vector_type(8)));
typedef int          v4i __attribute__((ext_vector_type(4)));

#define CN     128
#define NPIX   4096
#define HEADS  4
#define DH     32
#define BR     64
#define BC     64
#define HB     ((size_t)4 * HEADS * NPIX * DH)   // elements per q/k/v buffer

#if defined(__has_builtin)
#if __has_builtin(__builtin_amdgcn_tensor_load_to_lds) && \
    __has_builtin(__builtin_amdgcn_s_wait_tensorcnt)
#define HAVE_TDM 1
#endif
#endif

// ---------------- WMMA helpers (gfx1250 layouts, ISA 7.12.2) ----------------

__device__ __forceinline__ v8f wmma_f16(v16h a, v16h b, v8f c) {
  return __builtin_amdgcn_wmma_f32_16x16x32_f16(false, a, false, b, (short)0, c,
                                                false, false);
}

// A fragment (16x32, M x K) from row-major [m][k] f16 buffer with leading dim ld.
__device__ __forceinline__ v16h load_a_rm(const f16* A, int m0, int k0, int ld, int lane) {
  const f16* p = A + (size_t)(m0 + (lane & 15)) * ld + k0 + ((lane >> 4) << 4);
  v16h a;
#pragma unroll
  for (int e = 0; e < 8; ++e) a[e] = p[e];
#pragma unroll
  for (int e = 0; e < 8; ++e) a[8 + e] = p[8 + e];
  return a;
}

// B fragment (32x16, K x N) from K-major [k][n] buffer with leading dim ld.
__device__ __forceinline__ v16h load_b_km(const f16* B, int k0, int n0, int ld, int lane) {
  const f16* p = B + (size_t)(k0 + ((lane >> 4) << 4)) * ld + n0 + (lane & 15);
  v16h b;
#pragma unroll
  for (int e = 0; e < 16; ++e) b[e] = p[e * ld];
  return b;
}

// B fragment (32x16, K x N) from a row-major [n][k] buffer (B transposed);
// used for Q*K^T where the K-tile is stored [key][d]. 16 contiguous f16/lane.
__device__ __forceinline__ v16h load_b_nm(const f16* Bt, int k0, int n0, int ld, int lane) {
  const f16* p = Bt + (size_t)(n0 + (lane & 15)) * ld + k0 + ((lane >> 4) << 4);
  v16h b;
#pragma unroll
  for (int e = 0; e < 16; ++e) b[e] = p[e];
  return b;
}

#if defined(HAVE_TDM)
// ---------------- Tensor Data Mover: 4KB K/V tile -> LDS, with padding -------
// 1-D tile of 2048 f16; pad_enable inserts 4 DWORDs (16B) after every 16 DWORDs
// (64B) so the LDS image has row stride DH+8 f16 = 80B (bank-friendly).
__device__ __forceinline__ void tdm_load_tile(const f16* gsrc, void* lds_dst) {
  const unsigned long long ga = (unsigned long long)(uintptr_t)gsrc;
  const unsigned int       la = (unsigned int)(uintptr_t)lds_dst; // AS3 offset
  v4u g0;
  g0[0] = 1u;                                   // count=1, user descriptor
  g0[1] = la;                                   // lds_addr (bytes)
  g0[2] = (unsigned int)ga;                     // global_addr[31:0]
  g0[3] = (unsigned int)(ga >> 32) | (2u << 30);// global_addr[56:32] | type=2
  v8i g1;
  // data_size=1 (2B) @16 | pad_enable @20 | pad_interval=3 (16 DW) @22
  // | pad_amount=3 (4 DW) @25 ; workgroup_mask=0 (not in a cluster)
  g1[0] = (int)((1u << 16) | (1u << 20) | (3u << 22) | (3u << 25));
  const unsigned int tdim0 = NPIX * DH;         // tensor length (elements)
  g1[1] = (int)((tdim0 & 0xFFFFu) << 16);       // tensor_dim0[15:0] @ 63:48
  g1[2] = (int)(((tdim0 >> 16) & 0xFFFFu) | (1u << 16)); // dim0 hi | tensor_dim1=1
  g1[3] = (int)((unsigned)(BC * DH) << 16);     // tile_dim0 = 2048 @ 127:112
  g1[4] = 1;                                    // tile_dim1 = 1, tile_dim2 = 0
  g1[5] = (int)tdim0;                           // tensor_dim0_stride[31:0]
  g1[6] = 0;
  g1[7] = 0;
  v4i z4 = {0, 0, 0, 0};
#if __has_include(<hip/amd_detail/amd_gfx1250_TDM.h>)
  v8i z8 = {0, 0, 0, 0, 0, 0, 0, 0};            // clang-23 / therock: 6-arg form
  __builtin_amdgcn_tensor_load_to_lds(g0, g1, z4, z4, z8, 0);
#else
  __builtin_amdgcn_tensor_load_to_lds(g0, g1, z4, z4, 0); // ROCm 7.2: 5-arg form
#endif
}
#endif

// ------- Kernel 0: one-shot fp32 -> f16 weights; q-scale folded into w_qkv ---
__global__ __launch_bounds__(256) void wconv_kernel(
    const float* __restrict__ wq, const float* __restrict__ wo,
    f16* __restrict__ wqh, f16* __restrict__ woh)
{
  const float scale = 0.17677669529663687f; // 1/sqrt(32), applied to q rows
  int i = blockIdx.x * 256 + threadIdx.x;
  if (i < 3 * CN * CN) wqh[i] = (f16)(wq[i] * ((i < CN * CN) ? scale : 1.0f));
  if (i < CN * CN)     woh[i] = (f16)wo[i];
}

// ---------------- Kernel 1: channel LayerNorm + QKV projection ----------------
// grid = B * (NPIX/64), block = 256 (8 waves). qkv = contiguous q|k|v buffers.
__global__ __launch_bounds__(256) void ln_qkv_kernel(
    const float* __restrict__ x, const float* __restrict__ g, const float* __restrict__ bchan,
    const f16* __restrict__ wqh, f16* __restrict__ qkv)
{
  __shared__ float xs[CN][66];     // x tile, padded vs bank conflicts
  __shared__ float meanr[64], rstdr[64];
  __shared__ f16 xn[CN][72];       // normalized tile as GEMM B matrix [c][p]

  const int tid  = threadIdx.x;
  const int lane = tid & 31;
  const int wave = tid >> 5;
  const int bidx = blockIdx.x >> 6;        // batch
  const int p0   = (blockIdx.x & 63) * 64; // pixel tile origin

  const float* xb = x + (size_t)bidx * CN * NPIX + p0;
#pragma unroll 4
  for (int i = tid; i < CN * 64; i += 256) {
    int c = i >> 6, p = i & 63;
    xs[c][p] = xb[(size_t)c * NPIX + p];
  }
  __syncthreads();

  if (tid < 64) {
    float s = 0.f, s2 = 0.f;
#pragma unroll 8
    for (int c = 0; c < CN; ++c) { float v = xs[c][tid]; s += v; s2 += v * v; }
    float m   = s * (1.0f / CN);
    float var = s2 * (1.0f / CN) - m * m;
    meanr[tid] = m;
    rstdr[tid] = __frsqrt_rn(var + 1e-5f);
  }
  __syncthreads();

#pragma unroll 4
  for (int i = tid; i < CN * 64; i += 256) {
    int c = i >> 6, p = i & 63;
    float v = (xs[c][p] - meanr[p]) * rstdr[p] * g[c] + bchan[c];
    xn[c][p] = (f16)v;
  }
  __syncthreads();

  // GEMM: (384 x 128) * (128 x 64) -> 24 Mtiles x 4 Ntiles = 96 WMMA tiles
  for (int tt = wave; tt < 96; tt += 8) {
    int mt = tt >> 2, nt = tt & 3;
    v8f acc = {};
#pragma unroll
    for (int kk = 0; kk < CN; kk += 32) {
      v16h a = load_a_rm(wqh, mt * 16, kk, CN, lane);  // f16 weights, L2-resident
      v16h b = load_b_km(&xn[0][0], kk, nt * 16, 72, lane);
      acc = wmma_f16(a, b, acc);
    }
    // C layout: row = o0 + r, col = pixel; 8 consecutive o share head & q/k/v.
    int p   = p0 + nt * 16 + (lane & 15);
    int o0  = mt * 16 + ((lane >> 4) << 3);
    int sel = o0 >> 7;          // 0 = q, 1 = k, 2 = v  (contiguous slabs)
    int om  = o0 & 127;
    int hd  = om >> 5;
    int d0  = om & 31;
    v8h pk;
#pragma unroll
    for (int r = 0; r < 8; ++r) pk[r] = (f16)acc[r];   // scale pre-folded
    f16* dst = qkv + (size_t)sel * HB +
               ((((size_t)bidx * HEADS + hd) * NPIX + p) << 5) + d0;
    *reinterpret_cast<v8h*>(dst) = pk;   // 16B packed store of 8 d-values
  }
}

// ---------------- Kernel 2: flash attention ----------------
// grid = B*HEADS*(NPIX/BR), block = 256 (8 waves)
__global__ __launch_bounds__(256) void attn_kernel(
    const f16* __restrict__ qw, const f16* __restrict__ kw, const f16* __restrict__ vw,
    f16* __restrict__ ow)
{
  __shared__ __align__(16) f16 qs[BR][DH + 8];
  __shared__ __align__(16) f16 ks[2][BC][DH + 8];   // double-buffered K tiles
  __shared__ __align__(16) f16 vs[2][BC][DH + 8];   // double-buffered V tiles
  __shared__ float ss[BR][BC + 1];                  // scores (fp32)
  __shared__ __align__(16) f16 ps[BR][BC + 8];      // exp(S - m) (f16)
  __shared__ float os[BR][DH + 8];                  // running O accumulator
  __shared__ float mrow[BR], lrow[BR], arow[BR];
  __shared__ float redm[BR][4], reds[BR][4];        // softmax partials

  const int tid  = threadIdx.x;
  const int lane = tid & 31;
  const int wave = tid >> 5;
  const int bh   = blockIdx.x >> 6;        // b*HEADS + h
  const int qt   = blockIdx.x & 63;        // query tile
  const size_t base = (size_t)bh * NPIX * DH;
  const int NJ   = NPIX / BC;

  const f16* Q = qw + base + (size_t)qt * BR * DH;
#pragma unroll 4
  for (int i = tid; i < BR * DH; i += 256) {
    qs[i >> 5][i & 31] = Q[i];
    os[i >> 5][i & 31] = 0.0f;
  }
  if (tid < BR) { mrow[tid] = -1e30f; lrow[tid] = 0.0f; }

#if defined(HAVE_TDM)
  if (wave == 0) {  // DMA tile 0 into buffer 0 (TDM ignores EXEC; one op/wave)
    tdm_load_tile(kw + base, &ks[0][0][0]);
    tdm_load_tile(vw + base, &vs[0][0][0]);
  }
#endif
  __syncthreads();

  for (int j = 0; j < NJ; ++j) {
#if defined(HAVE_TDM)
    const int cur = j & 1;
    if (wave == 0) __builtin_amdgcn_s_wait_tensorcnt(0);  // current tiles landed
    __syncthreads();
    if (wave == 0 && j + 1 < NJ) {  // overlap next DMA with this tile's math
      tdm_load_tile(kw + base + (size_t)(j + 1) * BC * DH, &ks[cur ^ 1][0][0]);
      tdm_load_tile(vw + base + (size_t)(j + 1) * BC * DH, &vs[cur ^ 1][0][0]);
    }
#else
    const int cur = 0;
    const f16* K = kw + base + (size_t)j * BC * DH;
    const f16* V = vw + base + (size_t)j * BC * DH;
#pragma unroll 4
    for (int i = tid; i < BC * DH; i += 256) {
      ks[0][i >> 5][i & 31] = K[i];
      vs[0][i >> 5][i & 31] = V[i];
    }
    if (j + 1 < NJ) {  // global_prefetch_b8 of next K/V tile
      __builtin_prefetch(K + BC * DH + (tid << 3), 0, 1);
      __builtin_prefetch(V + BC * DH + (tid << 3), 0, 1);
    }
    __syncthreads();
#endif
    const f16* ksb = &ks[cur][0][0];
    const f16* vsb = &vs[cur][0][0];

    // S = Q * K^T : 4x4 = 16 tiles of 16x16, K-dim = 32 -> 1 WMMA each
    for (int tt = wave; tt < 16; tt += 8) {
      int mt = tt >> 2, nt = tt & 3;
      v16h a = load_a_rm(&qs[0][0], mt * 16, 0, DH + 8, lane);
      v16h b = load_b_nm(ksb, 0, nt * 16, DH + 8, lane);
      v8f acc = {};
      acc = wmma_f16(a, b, acc);
      int col = nt * 16 + (lane & 15);
      int r0  = mt * 16 + ((lane >> 4) << 3);
#pragma unroll
      for (int r = 0; r < 8; ++r) ss[r0 + r][col] = acc[r];
    }
    __syncthreads();

    // online softmax, 4 threads per row (16 columns each)
    {
      const int r = tid >> 2, qd = tid & 3;
      const float* srow = &ss[r][qd * 16];
      float mx = -1e30f;
#pragma unroll
      for (int c = 0; c < 16; ++c) mx = fmaxf(mx, srow[c]);
      redm[r][qd] = mx;
    }
    __syncthreads();
    {
      const int r = tid >> 2, qd = tid & 3;
      float mx = fmaxf(fmaxf(redm[r][0], redm[r][1]),
                       fmaxf(redm[r][2], redm[r][3]));
      mx = fmaxf(mx, mrow[r]);
      const float* srow = &ss[r][qd * 16];
      f16* prow = &ps[r][qd * 16];
      float s = 0.f;
#pragma unroll
      for (int c = 0; c < 16; ++c) {
        float e = __expf(srow[c] - mx);
        prow[c] = (f16)e;
        s += e;
      }
      reds[r][qd] = s;
    }
    __syncthreads();
    if (tid < BR) {  // single writer per row finalizes the running stats
      float mx = fmaxf(fmaxf(redm[tid][0], redm[tid][1]),
                       fmaxf(redm[tid][2], redm[tid][3]));
      mx = fmaxf(mx, mrow[tid]);
      float al = __expf(mrow[tid] - mx);
      float s  = reds[tid][0] + reds[tid][1] + reds[tid][2] + reds[tid][3];
      mrow[tid] = mx;
      lrow[tid] = lrow[tid] * al + s;
      arow[tid] = al;
    }
    __syncthreads();

    // rescale running O by alpha
#pragma unroll 4
    for (int i = tid; i < BR * DH; i += 256) {
      int r = i >> 5, d = i & 31;
      os[r][d] *= arow[r];
    }
    __syncthreads();

    // O += P * V : 4x2 = 8 tiles of 16x16, K-dim = 64 -> 2 WMMA each (1/wave)
    {
      int mt = wave >> 1, nt = wave & 1;
      int col = nt * 16 + (lane & 15);
      int r0  = mt * 16 + ((lane >> 4) << 3);
      v8f acc;
#pragma unroll
      for (int r = 0; r < 8; ++r) acc[r] = os[r0 + r][col];
#pragma unroll
      for (int kk = 0; kk < BC; kk += 32) {
        v16h a = load_a_rm(&ps[0][0], mt * 16, kk, BC + 8, lane);
        v16h b = load_b_km(vsb, kk, nt * 16, DH + 8, lane);
        acc = wmma_f16(a, b, acc);
      }
#pragma unroll
      for (int r = 0; r < 8; ++r) os[r0 + r][col] = acc[r];
    }
    __syncthreads();
  }

  // epilogue: O / l, write as [b][c=h*32+d][p] f16 for the projection GEMM
  const int hd = bh & (HEADS - 1);
  const int bidx = bh >> 2;
#pragma unroll 4
  for (int i = tid; i < BR * DH; i += 256) {
    int r = i >> 5, d = i & 31;
    float v = os[r][d] / lrow[r];
    int p = qt * BR + r;
    ow[((size_t)bidx * CN + (hd * DH + d)) * NPIX + p] = (f16)v;
  }
}

// ---------------- Kernel 3: output projection + bias ----------------
// grid = B * (NPIX/64), block = 256 (8 waves)
__global__ __launch_bounds__(256) void proj_kernel(
    const f16* __restrict__ ow, const f16* __restrict__ woh,
    const float* __restrict__ b_out, float* __restrict__ out)
{
  __shared__ f16   bs[CN][72];      // attention-out tile as B matrix [c][p]
  __shared__ float ts[8][16][17];   // per-wave C staging for coalesced stores

  const int tid  = threadIdx.x;
  const int lane = tid & 31;
  const int wave = tid >> 5;
  const int bidx = blockIdx.x >> 6;
  const int p0   = (blockIdx.x & 63) * 64;

  const f16* src = ow + (size_t)bidx * CN * NPIX + p0;
#pragma unroll 4
  for (int i = tid; i < CN * 64; i += 256) {
    int c = i >> 6, p = i & 63;
    bs[c][p] = src[(size_t)c * NPIX + p];
  }
  __syncthreads();

  // (128 x 128) * (128 x 64) -> 8 Mtiles x 4 Ntiles = 32 tiles, 4 per wave
  for (int tt = wave; tt < 32; tt += 8) {
    int mt = tt >> 2, nt = tt & 3;
    v8f acc = {};
#pragma unroll
    for (int kk = 0; kk < CN; kk += 32) {
      v16h a = load_a_rm(woh, mt * 16, kk, CN, lane);  // f16 weights, L2-resident
      v16h b = load_b_km(&bs[0][0], kk, nt * 16, 72, lane);
      acc = wmma_f16(a, b, acc);
    }
    int col = lane & 15;
    int r0  = (lane >> 4) << 3;
#pragma unroll
    for (int r = 0; r < 8; ++r) ts[wave][r0 + r][col] = acc[r];
    __syncthreads();   // uniform: every wave runs exactly 4 iterations

    int row = lane >> 1, ch = (lane & 1) << 3;
    int o = mt * 16 + row;
    float bo = b_out[o];
    float* dst = out + ((size_t)bidx * CN + o) * NPIX + p0 + nt * 16 + ch;
#pragma unroll
    for (int e = 0; e < 8; ++e) dst[e] = ts[wave][row][ch + e] + bo;
    __syncthreads();
  }
}

// ---------------- launcher ----------------
extern "C" void kernel_launch(void* const* d_in, const int* in_sizes, int n_in,
                              void* d_out, int out_size, void* d_ws, size_t ws_size,
                              hipStream_t stream) {
  (void)in_sizes; (void)n_in; (void)out_size; (void)ws_size;
  const float* x     = (const float*)d_in[0];
  const float* g     = (const float*)d_in[1];
  const float* bchan = (const float*)d_in[2];
  const float* w_qkv = (const float*)d_in[3];
  const float* w_out = (const float*)d_in[4];
  const float* b_out = (const float*)d_in[5];
  float* out = (float*)d_out;

  f16* qw  = (f16*)d_ws;          // q | k | v contiguous slabs of HB elements
  f16* kw  = qw + HB;
  f16* vw  = kw + HB;
  f16* ow  = vw + HB;             // [B][128][NPIX] f16
  f16* wqh = ow + HB;             // 384x128 f16 weights (q rows pre-scaled)
  f16* woh = wqh + 3 * CN * CN;   // 128x128 f16 weights

  wconv_kernel<<<(3 * CN * CN + 255) / 256, 256, 0, stream>>>(w_qkv, w_out, wqh, woh);
  ln_qkv_kernel<<<4 * (NPIX / 64), 256, 0, stream>>>(x, g, bchan, wqh, qw);
  attn_kernel<<<4 * HEADS * (NPIX / BR), 256, 0, stream>>>(qw, kw, vw, ow);
  proj_kernel<<<4 * (NPIX / 64), 256, 0, stream>>>(ow, woh, b_out, out);
}
